// HeteroGNN_41635412968139
// MI455X (gfx1250) — matrix operand order repaired and checked by reading, not compiled, not run on hardware
//
#include <hip/hip_runtime.h>

typedef __attribute__((ext_vector_type(16))) _Float16 v16h;
typedef __attribute__((ext_vector_type(8)))  _Float16 v8h;
typedef __attribute__((ext_vector_type(4)))  _Float16 v4h;
typedef __attribute__((ext_vector_type(8)))  float    v8f;

#define DCH 128

// ---------------------------------------------------------------------------
// Utility fills / conversions
// ---------------------------------------------------------------------------
__global__ void zero_i32(int* __restrict__ p, size_t n) {
    size_t i = (size_t)blockIdx.x * blockDim.x + threadIdx.x;
    if (i < n) p[i] = 0;
}

// x (f32) -> xh (f16), 4 elements / thread
__global__ void cvt_f16(const float* __restrict__ x, _Float16* __restrict__ xh,
                        size_t n4) {
    size_t i = (size_t)blockIdx.x * blockDim.x + threadIdx.x;
    if (i >= n4) return;
    float4 v = *(const float4*)(x + i * 4);
    v4h o;
    o[0] = (_Float16)v.x; o[1] = (_Float16)v.y;
    o[2] = (_Float16)v.z; o[3] = (_Float16)v.w;
    *(v4h*)(xh + i * 4) = o;
}

// ---------------------------------------------------------------------------
// Pack a 128x128 f32 weight into WMMA f16 B-fragment layout:
//   P[(((kc*8 + nt)*32) + lane)*16 + i] = f16( W[(kc*32 + (lane<16?0:8) + (i<8?i:i+8))*128
//                                              + nt*16 + (lane&15)] )
// so each lane of a wave loads its 16-element fragment as 32 contiguous bytes.
// ---------------------------------------------------------------------------
__global__ void pack_weights(const float* __restrict__ W, _Float16* __restrict__ P) {
    int t    = blockIdx.x * blockDim.x + threadIdx.x;   // 0..16383
    int i    = t & 15;
    int lane = (t >> 4) & 31;
    int nt   = (t >> 9) & 7;
    int kc   = (t >> 12) & 3;
    int n    = nt * 16 + (lane & 15);
    int K    = kc * 32 + ((lane < 16) ? 0 : 8) + ((i < 8) ? i : i + 8);
    P[t] = (_Float16)W[K * DCH + n];
}

// ---------------------------------------------------------------------------
// CSR build: degree histogram, single-block exclusive scan, bucket fill
// ---------------------------------------------------------------------------
__global__ void hist_deg(const int* __restrict__ ei, int* __restrict__ deg, int E) {
    int e = blockIdx.x * blockDim.x + threadIdx.x;
    if (e < E) atomicAdd(&deg[ei[E + e]], 1);
}

__global__ __launch_bounds__(256) void scan_excl(const int* __restrict__ deg,
                                                 int* __restrict__ off, int N) {
    __shared__ int buf[256];
    __shared__ int carry;
    if (threadIdx.x == 0) carry = 0;
    __syncthreads();
    for (int base = 0; base < N; base += 256) {
        int i = base + threadIdx.x;
        int v = (i < N) ? deg[i] : 0;
        buf[threadIdx.x] = v;
        __syncthreads();
        for (int o = 1; o < 256; o <<= 1) {        // inclusive Hillis-Steele
            int t = (threadIdx.x >= o) ? buf[threadIdx.x - o] : 0;
            __syncthreads();
            buf[threadIdx.x] += t;
            __syncthreads();
        }
        if (i < N) off[i] = carry + buf[threadIdx.x] - v;   // exclusive
        __syncthreads();
        if (threadIdx.x == 255) carry += buf[255];
        __syncthreads();
    }
}

__global__ void fill_csr(const int* __restrict__ ei, const int* __restrict__ off,
                         int* __restrict__ cursor, int* __restrict__ srclist, int E) {
    int e = blockIdx.x * blockDim.x + threadIdx.x;
    if (e >= E) return;
    int tgt = ei[E + e];
    int p = atomicAdd(&cursor[tgt], 1);
    srclist[off[tgt] + p] = ei[e];
}

// ---------------------------------------------------------------------------
// Node-centric mean aggregation over the fp16 feature table: one wave per
// node, lane owns 4 channels (b64 gather). src ids loaded coalesced and
// broadcast via shfl; f32 register accumulation; single fp16 store of the
// mean. Lanes prefetch the rows they just fetched ids for.
// ---------------------------------------------------------------------------
__global__ __launch_bounds__(256) void aggregate_mean(const _Float16* __restrict__ xh,
                                                      const int* __restrict__ off,
                                                      const int* __restrict__ deg,
                                                      const int* __restrict__ srclist,
                                                      _Float16* __restrict__ aggh, int N) {
    int t    = blockIdx.x * blockDim.x + threadIdx.x;
    int node = t >> 5;
    int lane = t & 31;
    if (node >= N) return;
    int base = off[node];
    int d    = deg[node];
    float a0 = 0.f, a1 = 0.f, a2 = 0.f, a3 = 0.f;
    for (int j0 = 0; j0 < d; j0 += 32) {
        int sj = (j0 + lane < d) ? srclist[base + j0 + lane] : 0;
        if (j0 + lane < d)
            __builtin_prefetch(xh + (size_t)sj * DCH, 0, 0);   // global_prefetch_b8
        int lim = min(32, d - j0);
        for (int k = 0; k < lim; ++k) {
            int s = __shfl(sj, k, 32);
            const v4h v = *(const v4h*)(xh + (size_t)s * DCH + lane * 4);
            a0 += (float)v[0]; a1 += (float)v[1];
            a2 += (float)v[2]; a3 += (float)v[3];
        }
    }
    const float inv = 1.0f / fmaxf((float)d, 1.0f);
    v4h o;
    o[0] = (_Float16)(a0 * inv); o[1] = (_Float16)(a1 * inv);
    o[2] = (_Float16)(a2 * inv); o[3] = (_Float16)(a3 * inv);
    *(v4h*)(aggh + (size_t)node * DCH + lane * 4) = o;
}

// ---------------------------------------------------------------------------
// WMMA fragment helpers (16-bit A 16x32 layout / packed B layout)
// ---------------------------------------------------------------------------
__device__ __forceinline__ v16h lds_a_frag(const _Float16* rowbase, int koff) {
    v8h lo = *(const v8h*)(rowbase + koff);
    v8h hi = *(const v8h*)(rowbase + koff + 16);
    v16h r;
#pragma unroll
    for (int i = 0; i < 8; ++i) { r[i] = lo[i]; r[i + 8] = hi[i]; }
    return r;
}

__device__ __forceinline__ v8f wmma_f16(v16h a, v16h b, v8f c) {
    return __builtin_amdgcn_wmma_f32_16x16x32_f16(false, a, false, b,
                                                  (short)0, c, false, false);
}

// ---------------------------------------------------------------------------
// Fused node kernel: 16 rows per block, 8 waves = 8 column tiles.
//   h      = agg@Wl + bl + x@Wr        (8 WMMA / wave)
//   logits = h@Wp + bp                 (4 WMMA / wave)
//   out   += 0.5 * softmax(logits)
// Inputs are already fp16, so phase 1 is a pure b128 copy into LDS.
// ---------------------------------------------------------------------------
__global__ __launch_bounds__(256) void fused_gnn(const _Float16* __restrict__ xh,
                                                 const _Float16* __restrict__ aggh,
                                                 const _Float16* __restrict__ PWl,
                                                 const _Float16* __restrict__ PWr,
                                                 const _Float16* __restrict__ PWp,
                                                 const float* __restrict__ bl,
                                                 const float* __restrict__ bp,
                                                 float* __restrict__ out, int firstPass) {
    __shared__ _Float16 A1[16 * DCH];   // mean-aggregated neighbors
    __shared__ _Float16 A2[16 * DCH];   // self features
    __shared__ _Float16 Hs[16 * DCH];   // hidden h
    __shared__ float    Ls[16 * DCH];   // logits (f32)

    const int t  = threadIdx.x;
    const int r0 = blockIdx.x * 16;

    // ---- phase 1: stage agg (already mean, f16) and x rows ----
    {
        const int row = t >> 4;            // 0..15
        const int c0  = (t & 15) * 8;      // 0..120 (8 f16 = 16B per thread)
        const size_t g = (size_t)(r0 + row) * DCH + c0;
        uint4 a = *(const uint4*)(aggh + g);
        uint4 b = *(const uint4*)(xh + g);
        *(uint4*)(A1 + row * DCH + c0) = a;
        *(uint4*)(A2 + row * DCH + c0) = b;
    }
    __syncthreads();

    const int wave = t >> 5;               // column tile 0..7
    const int lane = t & 31;
    const int m    = lane & 15;
    const int hi16 = lane >> 4;            // 0 or 1
    const int kb   = hi16 * 8;             // A-fragment K base inside 32-chunk
    const int ncol = wave * 16 + m;        // global output column of this lane

    // ---- phase 2: h = A1@Wl + A2@Wr + bl ----
    v8f acc = {};
    const _Float16* a1row = A1 + m * DCH;
    const _Float16* a2row = A2 + m * DCH;
#pragma unroll
    for (int kc = 0; kc < 4; ++kc) {
        v16h af = lds_a_frag(a1row, kc * 32 + kb);
        v16h wf = *(const v16h*)(PWl + (((kc * 8 + wave) * 32) + lane) * 16);
        acc = wmma_f16(af, wf, acc);
    }
#pragma unroll
    for (int kc = 0; kc < 4; ++kc) {
        v16h af = lds_a_frag(a2row, kc * 32 + kb);
        v16h wf = *(const v16h*)(PWr + (((kc * 8 + wave) * 32) + lane) * 16);
        acc = wmma_f16(af, wf, acc);
    }
    const float biasl = bl[ncol];
#pragma unroll
    for (int v = 0; v < 8; ++v) {          // C/D layout: row = v + hi16*8, col = ncol
        Hs[(v + hi16 * 8) * DCH + ncol] = (_Float16)(acc[v] + biasl);
    }
    __syncthreads();

    // ---- phase 3: logits = h@Wp + bp ----
    v8f acc2 = {};
    const _Float16* hrow = Hs + m * DCH;
#pragma unroll
    for (int kc = 0; kc < 4; ++kc) {
        v16h af = lds_a_frag(hrow, kc * 32 + kb);
        v16h wf = *(const v16h*)(PWp + (((kc * 8 + wave) * 32) + lane) * 16);
        acc2 = wmma_f16(af, wf, acc2);
    }
    const float biasp = bp[ncol];
#pragma unroll
    for (int v = 0; v < 8; ++v) {
        Ls[(v + hi16 * 8) * DCH + ncol] = acc2[v] + biasp;
    }
    __syncthreads();

    // ---- phase 4: row softmax (wave handles rows 2*wave, 2*wave+1) ----
#pragma unroll
    for (int rr = 0; rr < 2; ++rr) {
        const int row = wave * 2 + rr;
        float4 v4 = *(const float4*)(Ls + row * DCH + lane * 4);
        float mx = fmaxf(fmaxf(v4.x, v4.y), fmaxf(v4.z, v4.w));
#pragma unroll
        for (int off = 16; off > 0; off >>= 1)
            mx = fmaxf(mx, __shfl_xor(mx, off, 32));
        float e0 = __expf(v4.x - mx), e1 = __expf(v4.y - mx);
        float e2 = __expf(v4.z - mx), e3 = __expf(v4.w - mx);
        float s = e0 + e1 + e2 + e3;
#pragma unroll
        for (int off = 16; off > 0; off >>= 1)
            s += __shfl_xor(s, off, 32);
        const float sc = 0.5f / s;
        float4 o;
        o.x = e0 * sc; o.y = e1 * sc; o.z = e2 * sc; o.w = e3 * sc;
        float* op = out + (size_t)(r0 + row) * DCH + lane * 4;
        if (!firstPass) {
            float4 prev = *(const float4*)op;
            o.x += prev.x; o.y += prev.y; o.z += prev.z; o.w += prev.w;
        }
        *(float4*)op = o;
    }
}

// ---------------------------------------------------------------------------
// Launch
// ---------------------------------------------------------------------------
extern "C" void kernel_launch(void* const* d_in, const int* in_sizes, int n_in,
                              void* d_out, int out_size, void* d_ws, size_t ws_size,
                              hipStream_t stream) {
    const float* x  = (const float*)d_in[0];
    const float* Wl = (const float*)d_in[1];
    const float* bl = (const float*)d_in[2];
    const float* Wr = (const float*)d_in[3];
    const float* Wp = (const float*)d_in[4];
    const float* bp = (const float*)d_in[5];
    const int* eia  = (const int*)d_in[6];
    const int* eib  = (const int*)d_in[7];
    float* out      = (float*)d_out;

    const int N = in_sizes[0] / DCH;   // 50000 (multiple of 16)
    const int E = in_sizes[6] / 2;     // 800000

    // workspace layout:
    //   xh [N*128 f16] | aggh [N*128 f16] | deg [N i32] | cursor [N i32]
    //   | off [N i32] | srclist [E i32] | PWl|PWr|PWp [128*128 f16 each]
    char* ws        = (char*)d_ws;
    _Float16* xh    = (_Float16*)ws;
    size_t o        = (size_t)N * DCH * sizeof(_Float16);
    _Float16* aggh  = (_Float16*)(ws + o);  o += (size_t)N * DCH * sizeof(_Float16);
    int* deg        = (int*)(ws + o);       o += (size_t)N * sizeof(int);
    int* cursor     = (int*)(ws + o);       o += (size_t)N * sizeof(int);
    int* off        = (int*)(ws + o);       o += (size_t)N * sizeof(int);
    int* srclist    = (int*)(ws + o);       o += (size_t)E * sizeof(int);
    o = (o + 31) & ~(size_t)31;
    _Float16* PWl   = (_Float16*)(ws + o);
    _Float16* PWr   = PWl + DCH * DCH;
    _Float16* PWp   = PWr + DCH * DCH;

    // one-time prep
    const size_t n4 = (size_t)N * DCH / 4;
    cvt_f16<<<(int)((n4 + 255) / 256), 256, 0, stream>>>(x, xh, n4);
    const int packBlocks = (DCH * DCH) / 256;                 // 64
    pack_weights<<<packBlocks, 256, 0, stream>>>(Wl, PWl);
    pack_weights<<<packBlocks, 256, 0, stream>>>(Wr, PWr);
    pack_weights<<<packBlocks, 256, 0, stream>>>(Wp, PWp);

    const size_t zn  = (size_t)2 * N;                         // deg + cursor contiguous
    const int zBlocks = (int)((zn + 255) / 256);
    const int eBlocks = (E + 255) / 256;
    const int aBlocks = (N * 32 + 255) / 256;
    const int fBlocks = N / 16;

    const int* eis[2] = {eia, eib};
    for (int pass = 0; pass < 2; ++pass) {
        const int* ei = eis[pass];
        zero_i32<<<zBlocks, 256, 0, stream>>>(deg, zn);
        hist_deg<<<eBlocks, 256, 0, stream>>>(ei, deg, E);
        scan_excl<<<1, 256, 0, stream>>>(deg, off, N);
        fill_csr<<<eBlocks, 256, 0, stream>>>(ei, off, cursor, srclist, E);
        aggregate_mean<<<aBlocks, 256, 0, stream>>>(xh, off, deg, srclist, aggh, N);
        fused_gnn<<<fBlocks, 256, 0, stream>>>(xh, aggh, PWl, PWr, PWp, bl, bp, out,
                                               pass == 0 ? 1 : 0);
    }
}
